// CausalSelfAttention_38920993636989
// MI455X (gfx1250) — matrix operand order repaired and checked by reading, not compiled, not run on hardware
//
#include <hip/hip_runtime.h>

// ---------------------------------------------------------------------------
// CDNA5 (gfx1250) causal self-attention: bf16 WMMA GEMMs + flash attention.
// Wave32; all matmuls via v_wmma_f32_16x16x32_bf16.
// ---------------------------------------------------------------------------

typedef __bf16 bf16;
typedef __attribute__((ext_vector_type(8)))  __bf16 bf16x8;
typedef __attribute__((ext_vector_type(16))) __bf16 v16bf;
typedef __attribute__((ext_vector_type(8)))  float  v8f;

#define B_    2
#define S_    2048
#define D_    2048
#define H_    16
#define KV_   4
#define HD_   128
#define KVD_  512   // KV_*HD_

// RNE float -> bf16
__device__ __forceinline__ bf16 to_bf16(float f) {
  unsigned int u = __float_as_uint(f);
  u += 0x7FFFu + ((u >> 16) & 1u);
  unsigned short hs = (unsigned short)(u >> 16);
  bf16 r;
  __builtin_memcpy(&r, &hs, sizeof(r));
  return r;
}

__device__ __forceinline__ v16bf cat16(bf16x8 lo, bf16x8 hi) {
  return __builtin_shufflevector(lo, hi, 0,1,2,3,4,5,6,7,8,9,10,11,12,13,14,15);
}

// A-matrix fragment (16x32 bf16, MxK). ISA layout:
//   lanes 0-15 : M=lane,    K = 0..7 then 16..23
//   lanes 16-31: M=lane-16, K = 8..15 then 24..31
__device__ __forceinline__ v16bf load_a_frag(const bf16* A, int lda, int m0, int k0, int lane) {
  int ln = lane & 15, hf = lane >> 4;
  const bf16* p = A + (size_t)(m0 + ln) * lda + k0 + hf * 8;
  bf16x8 lo = *(const bf16x8*)p;
  bf16x8 hi = *(const bf16x8*)(p + 16);
  return cat16(lo, hi);
}

// B-matrix fragment (32x16 bf16, KxN) where memory holds B^T rows (N x K, K
// contiguous) -> lane n of half hf loads K = hf*8..hf*8+7 then 16+hf*8..16+hf*8+7 of row n0+n.
__device__ __forceinline__ v16bf load_b_frag(const bf16* W, int ldw, int n0, int k0, int lane) {
  int ln = lane & 15, hf = lane >> 4;
  const bf16* p = W + (size_t)(n0 + ln) * ldw + k0 + hf * 8;
  bf16x8 lo = *(const bf16x8*)p;
  bf16x8 hi = *(const bf16x8*)(p + 16);
  return cat16(lo, hi);
}

__device__ __forceinline__ v8f wmma_bf16(v16bf a, v16bf b, v8f c) {
  return __builtin_amdgcn_wmma_f32_16x16x32_bf16(false, a, false, b, (short)0, c, false, false);
}

// ---------------------------------------------------------------------------
// fp32 -> bf16 elementwise
// ---------------------------------------------------------------------------
__global__ void f32_to_bf16_kernel(const float* __restrict__ src,
                                   bf16* __restrict__ dst, size_t n) {
  size_t i = (size_t)blockIdx.x * 256 + threadIdx.x;
  if (i < n) dst[i] = to_bf16(src[i]);
}

// ---------------------------------------------------------------------------
// NT GEMM: C[m,n] = sum_k A[m,k] * W[n,k], A (MxK) bf16, W (NxK) bf16, C f32.
// 8 waves / block; each wave a 64x32 output block (4 m-tiles x 2 n-tiles):
// B fragments reused 4x, A fragments reused 2x -> 1.5 b128 loads per WMMA.
// ---------------------------------------------------------------------------
__global__ __launch_bounds__(256) void gemm_nt_kernel(
    const bf16* __restrict__ A, const bf16* __restrict__ W,
    float* __restrict__ C, int M, int N, int K) {
  int lane = threadIdx.x & 31;
  int wave = threadIdx.x >> 5;
  int m0 = (blockIdx.y * 8 + wave) * 64;
  int n0 = blockIdx.x * 32;
  if (m0 >= M) return;
  int ln = lane & 15;

  v8f acc[4][2];
#pragma unroll
  for (int i = 0; i < 4; ++i)
#pragma unroll
    for (int j = 0; j < 2; ++j)
      acc[i][j] = (v8f){0.f,0.f,0.f,0.f,0.f,0.f,0.f,0.f};

  for (int k0 = 0; k0 < K; k0 += 32) {
    // prefetch 16 k-steps ahead (speculative; dropped if OOB)
    __builtin_prefetch((const void*)(W + (size_t)(n0 + ln) * K + k0 + 512), 0, 0);
    __builtin_prefetch((const void*)(A + (size_t)(m0 + ln) * K + k0 + 512), 0, 0);
    __builtin_prefetch((const void*)(A + (size_t)(m0 + 32 + ln) * K + k0 + 512), 0, 0);

    v16bf b0 = load_b_frag(W, K, n0,      k0, lane);
    v16bf b1 = load_b_frag(W, K, n0 + 16, k0, lane);
#pragma unroll
    for (int i = 0; i < 4; ++i) {
      v16bf a = load_a_frag(A, K, m0 + i * 16, k0, lane);
      acc[i][0] = wmma_bf16(a, b0, acc[i][0]);
      acc[i][1] = wmma_bf16(a, b1, acc[i][1]);
    }
  }
  // C layout: VGPR r, lanes 0-15 -> (mt+r, n0+lane); lanes 16-31 -> (mt+8+r,..)
  int hf = lane >> 4;
#pragma unroll
  for (int i = 0; i < 4; ++i) {
    float* cp = C + (size_t)(m0 + i * 16 + hf * 8) * N + n0 + ln;
#pragma unroll
    for (int r = 0; r < 8; ++r) {
      cp[(size_t)r * N]      = acc[i][0][r];
      cp[(size_t)r * N + 16] = acc[i][1][r];
    }
  }
}

// ---------------------------------------------------------------------------
// RMS-norm + RoPE + gain (q,k) and V transpose, fp32 in -> bf16 out.
// block = 128 threads = one head-vector. blockIdx.y: [0,H) q, [H,H+KV) k,
// [H+KV,H+2KV) v (transpose only). sm_scale folded into Q.
// ---------------------------------------------------------------------------
__global__ __launch_bounds__(128) void normrope_kernel(
    const float* __restrict__ qf, const float* __restrict__ kf,
    const float* __restrict__ vf, const float* __restrict__ gain,
    bf16* __restrict__ Qb, bf16* __restrict__ Kb, bf16* __restrict__ Vt) {
  int bs = blockIdx.x;              // 0 .. B*S-1
  int b = bs >> 11, s = bs & (S_ - 1);
  int sel = blockIdx.y;
  int d = threadIdx.x;

  __shared__ float vec[HD_];
  __shared__ float red[4];

  float val;
  if (sel < H_)            val = qf[(size_t)bs * D_   + sel * HD_ + d];
  else if (sel < H_ + KV_) val = kf[(size_t)bs * KVD_ + (sel - H_) * HD_ + d];
  else                     val = vf[(size_t)bs * KVD_ + (sel - H_ - KV_) * HD_ + d];

  if (sel >= H_ + KV_) {   // V: just transpose to (B, KV, hd, S) bf16
    int kvh = sel - H_ - KV_;
    Vt[((size_t)(b * KV_ + kvh) * HD_ + d) * S_ + s] = to_bf16(val);
    return;
  }

  // RMS norm over 128 elements
  float sq = val * val;
#pragma unroll
  for (int off = 16; off; off >>= 1) sq += __shfl_xor(sq, off, 32);
  if ((threadIdx.x & 31) == 0) red[threadIdx.x >> 5] = sq;
  __syncthreads();
  float tot = red[0] + red[1] + red[2] + red[3];
  float inv = rsqrtf(tot * (1.0f / HD_) + 1.1920929e-7f);
  vec[d] = val * inv;
  __syncthreads();

  // RoPE over full head dim (half = 64)
  int i = d & 63;
  float x1 = vec[i], x2 = vec[i + 64];
  float freq = __expf(-((float)(2 * i) / (float)HD_) * 9.210340371976184f); // ln(10000)
  float ang = (float)s * freq;
  float c = cosf(ang), sn = sinf(ang);
  float out = (d < 64) ? (x1 * c + x2 * sn) : (-x1 * sn + x2 * c);

  if (sel < H_) {
    out *= gain[sel] * 0.08838834764831845f;   // q_gain * 1/sqrt(hd)
    Qb[(size_t)bs * D_ + sel * HD_ + d] = to_bf16(out);
  } else {
    Kb[(size_t)bs * KVD_ + (sel - H_) * HD_ + d] = to_bf16(out);
  }
}

// ---------------------------------------------------------------------------
// Flash attention, causal GQA. One wave (block of 32) per 32-row Q tile
// (two 16-row WMMA tiles sharing every K/V fragment -> 2 WMMAs per load).
// Q: (B,S,D) bf16 (scaled), K: (B,S,KVD) bf16, Vt: (B,KV,hd,S) bf16.
// ---------------------------------------------------------------------------
__global__ __launch_bounds__(32) void attn_kernel(
    const bf16* __restrict__ Qb, const bf16* __restrict__ Kb,
    const bf16* __restrict__ Vt, bf16* __restrict__ Yb) {
  int lane = threadIdx.x;
  int qt = blockIdx.x;          // 32-row q block
  int h  = blockIdx.y;
  int b  = blockIdx.z;
  int kvh = h >> 2;             // GQA group of 4
  int ln = lane & 15, hf = lane >> 4;

  __shared__ __align__(16) bf16 plds[2][16 * 32];  // P staging (C -> A frag)

  // Q fragments: 2 tiles x (16 x 128 over four k-steps of 32), loaded once
  const bf16* qbase = Qb + ((size_t)(b * S_ + qt * 32)) * D_ + h * HD_;
  v16bf qfrag[2][4];
#pragma unroll
  for (int qi = 0; qi < 2; ++qi)
#pragma unroll
    for (int j = 0; j < 4; ++j)
      qfrag[qi][j] = load_a_frag(qbase, D_, qi * 16, j * 32, lane);

  v8f O[2][8];
  float mstate[2][8], lstate[2][8];
#pragma unroll
  for (int qi = 0; qi < 2; ++qi) {
#pragma unroll
    for (int t = 0; t < 8; ++t) O[qi][t] = (v8f){0.f,0.f,0.f,0.f,0.f,0.f,0.f,0.f};
#pragma unroll
    for (int r = 0; r < 8; ++r) { mstate[qi][r] = -1e30f; lstate[qi][r] = 0.f; }
  }

  const bf16* kbase  = Kb + (size_t)(b * S_) * KVD_ + kvh * HD_;
  const bf16* vtbase = Vt + (size_t)(b * KV_ + kvh) * HD_ * S_;

  int lastk = qt * 32 + 31;
  for (int kt = 0; kt <= lastk; kt += 32) {
    // S = Q K^T for 32 keys x 32 q rows (K fragments shared by both q tiles)
    v8f s0[2], s1[2];
#pragma unroll
    for (int qi = 0; qi < 2; ++qi) {
      s0[qi] = (v8f){0.f,0.f,0.f,0.f,0.f,0.f,0.f,0.f};
      s1[qi] = (v8f){0.f,0.f,0.f,0.f,0.f,0.f,0.f,0.f};
    }
#pragma unroll
    for (int j = 0; j < 4; ++j) {
      v16bf kf0 = load_b_frag(kbase, KVD_, kt,      j * 32, lane);
      v16bf kf1 = load_b_frag(kbase, KVD_, kt + 16, j * 32, lane);
#pragma unroll
      for (int qi = 0; qi < 2; ++qi) {
        s0[qi] = wmma_bf16(qfrag[qi][j], kf0, s0[qi]);
        s1[qi] = wmma_bf16(qfrag[qi][j], kf1, s1[qi]);
      }
    }

    // online softmax; lane owns rows hf*8..hf*8+7 (column ln) of each tile
    __syncthreads();
#pragma unroll
    for (int qi = 0; qi < 2; ++qi) {
#pragma unroll
      for (int r = 0; r < 8; ++r) {
        int qr = qt * 32 + qi * 16 + hf * 8 + r;
        float v0 = (kt + ln      <= qr) ? s0[qi][r] : -1e30f;   // causal mask
        float v1 = (kt + 16 + ln <= qr) ? s1[qi][r] : -1e30f;
        float rm = fmaxf(v0, v1);
#pragma unroll
        for (int off = 1; off < 16; off <<= 1) rm = fmaxf(rm, __shfl_xor(rm, off, 32));
        float mnew  = fmaxf(mstate[qi][r], rm);
        float alpha = __expf(mstate[qi][r] - mnew);
        float p0 = __expf(v0 - mnew);
        float p1 = __expf(v1 - mnew);
        float rs = p0 + p1;
#pragma unroll
        for (int off = 1; off < 16; off <<= 1) rs += __shfl_xor(rs, off, 32);
        lstate[qi][r] = lstate[qi][r] * alpha + rs;
        mstate[qi][r] = mnew;
        plds[qi][(hf * 8 + r) * 32 + ln]      = to_bf16(p0);
        plds[qi][(hf * 8 + r) * 32 + 16 + ln] = to_bf16(p1);
#pragma unroll
        for (int t = 0; t < 8; ++t) O[qi][t][r] *= alpha;
      }
    }
    __syncthreads();

    // Re-layout P (C layout) -> A fragments via LDS
    v16bf pfrag[2];
#pragma unroll
    for (int qi = 0; qi < 2; ++qi) {
      bf16x8 plo = *(const bf16x8*)(&plds[qi][0] + ln * 32 + hf * 8);
      bf16x8 phi = *(const bf16x8*)(&plds[qi][0] + ln * 32 + 16 + hf * 8);
      pfrag[qi] = cat16(plo, phi);
    }

    // O += P @ V  (V fragments shared by both q tiles)
#pragma unroll
    for (int t = 0; t < 8; ++t) {
      v16bf vfr = load_b_frag(vtbase, S_, t * 16, kt, lane);
      O[0][t] = wmma_bf16(pfrag[0], vfr, O[0][t]);
      O[1][t] = wmma_bf16(pfrag[1], vfr, O[1][t]);
    }
  }

  // normalize and store Y (B,S,D) bf16
#pragma unroll
  for (int qi = 0; qi < 2; ++qi) {
    bf16* yp = Yb + ((size_t)(b * S_ + qt * 32 + qi * 16 + hf * 8)) * D_ + h * HD_ + ln;
#pragma unroll
    for (int r = 0; r < 8; ++r) {
      float invl = 1.0f / lstate[qi][r];
#pragma unroll
      for (int t = 0; t < 8; ++t)
        yp[(size_t)r * D_ + t * 16] = to_bf16(O[qi][t][r] * invl);
    }
  }
}

// ---------------------------------------------------------------------------
extern "C" void kernel_launch(void* const* d_in, const int* in_sizes, int n_in,
                              void* d_out, int out_size, void* d_ws, size_t ws_size,
                              hipStream_t stream) {
  (void)in_sizes; (void)n_in; (void)out_size; (void)ws_size;
  const float* x    = (const float*)d_in[0];
  const float* Wq   = (const float*)d_in[1];
  const float* Wk   = (const float*)d_in[2];
  const float* Wv   = (const float*)d_in[3];
  const float* Wp   = (const float*)d_in[4];
  const float* gain = (const float*)d_in[5];
  float* out = (float*)d_out;

  const size_t M = (size_t)B_ * S_;     // 4096 token rows
  char* ws = (char*)d_ws;
  size_t off = 0;
  auto take = [&](size_t bytes) -> char* {
    char* p = ws + off;
    off += (bytes + 255) & ~(size_t)255;
    return p;
  };

  bf16*  x_bf  = (bf16*)take(M * D_ * 2);            // reused later as Q bf16
  bf16*  wq_bf = (bf16*)take((size_t)D_ * D_ * 2);
  bf16*  wk_bf = (bf16*)take((size_t)KVD_ * D_ * 2);
  bf16*  wv_bf = (bf16*)take((size_t)KVD_ * D_ * 2);
  bf16*  wp_bf = (bf16*)take((size_t)D_ * D_ * 2);
  float* qf32  = (float*)take(M * D_ * 4);           // reused later as Y bf16
  float* kf32  = (float*)take(M * KVD_ * 4);
  float* vf32  = (float*)take(M * KVD_ * 4);
  bf16*  Kb    = (bf16*)take(M * KVD_ * 2);
  bf16*  Vt    = (bf16*)take(M * KVD_ * 2);
  bf16*  Qb    = x_bf;          // x_bf dead after v-GEMM
  bf16*  Yb    = (bf16*)qf32;   // qf32 dead after normrope

  auto cvt = [&](const float* s, bf16* d, size_t n) {
    f32_to_bf16_kernel<<<dim3((unsigned)((n + 255) / 256)), dim3(256), 0, stream>>>(s, d, n);
  };
  cvt(x,  x_bf,  M * D_);
  cvt(Wq, wq_bf, (size_t)D_ * D_);
  cvt(Wk, wk_bf, (size_t)KVD_ * D_);
  cvt(Wv, wv_bf, (size_t)KVD_ * D_);
  cvt(Wp, wp_bf, (size_t)D_ * D_);

  // QKV projections (each wave: 64x32 output block)
  gemm_nt_kernel<<<dim3(D_ / 32,   (unsigned)(M / 512)), dim3(256), 0, stream>>>(x_bf, wq_bf, qf32, (int)M, D_,   D_);
  gemm_nt_kernel<<<dim3(KVD_ / 32, (unsigned)(M / 512)), dim3(256), 0, stream>>>(x_bf, wk_bf, kf32, (int)M, KVD_, D_);
  gemm_nt_kernel<<<dim3(KVD_ / 32, (unsigned)(M / 512)), dim3(256), 0, stream>>>(x_bf, wv_bf, vf32, (int)M, KVD_, D_);

  // RMS norm + RoPE + gain (q,k), V transpose; fp32 -> bf16
  normrope_kernel<<<dim3((unsigned)M, H_ + 2 * KV_), dim3(128), 0, stream>>>(
      qf32, kf32, vf32, gain, Qb, Kb, Vt);

  // Flash attention (32 q rows per wave)
  attn_kernel<<<dim3(S_ / 32, H_, B_), dim3(32), 0, stream>>>(Qb, Kb, Vt, Yb);

  // Output projection -> fp32 d_out
  gemm_nt_kernel<<<dim3(D_ / 32, (unsigned)(M / 512)), dim3(256), 0, stream>>>(Yb, wp_bf, out, (int)M, D_, D_);
}
